// BaseLayerGate_20925080666297
// MI455X (gfx1250) — compile-verified, matrix operand order checked by reading, not run on hardware
//
#include <hip/hip_runtime.h>
#include <hip/hip_bf16.h>
#include <cstdint>

// Problem constants (from the reference).
#define D_MODEL    2048
#define NUM_EXPERT 16
#define N_TOKENS   16384                    // 4 * 4096
#define CAPACITY   (N_TOKENS / NUM_EXPERT)  // 1024 tokens per expert

typedef __attribute__((ext_vector_type(16))) __bf16   v16bf;
typedef __attribute__((ext_vector_type(8)))  float    v8f;
typedef __attribute__((ext_vector_type(8)))  unsigned v8u;
typedef __attribute__((ext_vector_type(4)))  float    f4;
typedef __attribute__((ext_vector_type(4)))  unsigned u4;

// Pack two fp32 into one dword of two bf16 (truncation) with a single
// v_perm_b32: dst = { hi.b3, hi.b2, lo.b3, lo.b2 }.
__device__ __forceinline__ unsigned pack_bf16(float lo, float hi) {
    return __builtin_amdgcn_perm(__builtin_bit_cast(unsigned, hi),
                                 __builtin_bit_cast(unsigned, lo),
                                 0x07060302u);
}

// ---------------------------------------------------------------------------
// Kernel A: one-time fp32 -> bf16 conversion of the centroids (16 x 2048).
// Output layout is plain row-major [E][D] bf16, which makes the WMMA B
// fragment two contiguous global_load_b128 per lane with no VALU work.
// ---------------------------------------------------------------------------
__global__ __launch_bounds__(256)
void centroid_bf16_kernel(const float* __restrict__ Cent,
                          unsigned* __restrict__ centbf /* dwords */) {
    const int i = blockIdx.x * blockDim.x + threadIdx.x;   // dword index
    if (i >= NUM_EXPERT * D_MODEL / 2) return;
    const float* p = Cent + 2 * (size_t)i;
    centbf[i] = pack_bf16(p[0], p[1]);
}

// ---------------------------------------------------------------------------
// Kernel B: affinities = X[16384,2048] @ Cent^T[2048,16] via WMMA bf16.
// One wave32 owns one 16x16 tile (16 tokens x all 16 experts).
// A fragment (16x32 bf16): lane L<16 -> row M=L, K = {k0..+7, k0+16..+23};
//                          lanes 16-31 -> same rows, K offsets +8.
// B fragment (32x16 bf16): lane holds one expert column, same K split;
// loaded directly as bf16 (pre-converted), 2x b128 per lane.
// ---------------------------------------------------------------------------
__global__ __launch_bounds__(256)
void affinity_wmma_kernel(const float* __restrict__ X,
                          const unsigned short* __restrict__ centbf,
                          float* __restrict__ aff) {
    const int lane   = threadIdx.x & 31;
    const int waveId = blockIdx.x * (blockDim.x >> 5) + (threadIdx.x >> 5);
    const int tok0   = waveId * 16;

    const int row  = lane & 15;            // token row (A) / expert col (B)
    const int koff = (lane & 16) ? 8 : 0;  // K sub-offset per half-wave

    const float*          arow = X + (size_t)(tok0 + row) * D_MODEL;
    const unsigned short* brow = centbf + (size_t)row * D_MODEL;

    v8f acc = {};
#pragma unroll 2
    for (int k0 = 0; k0 < D_MODEL; k0 += 32) {
        // Pull the next chunk of the streaming operand ahead of the loads.
        __builtin_prefetch(arow + k0 + 256, 0, 1);

        // A: 16 fp32 (two contiguous float4 pairs), packed to bf16 in-reg.
        f4 a0 = *(const f4*)(arow + k0 + koff);
        f4 a1 = *(const f4*)(arow + k0 + koff + 4);
        f4 a2 = *(const f4*)(arow + k0 + 16 + koff);
        f4 a3 = *(const f4*)(arow + k0 + 16 + koff + 4);

        v8u Au;
        Au[0] = pack_bf16(a0.x, a0.y);
        Au[1] = pack_bf16(a0.z, a0.w);
        Au[2] = pack_bf16(a1.x, a1.y);
        Au[3] = pack_bf16(a1.z, a1.w);
        Au[4] = pack_bf16(a2.x, a2.y);
        Au[5] = pack_bf16(a2.z, a2.w);
        Au[6] = pack_bf16(a3.x, a3.y);
        Au[7] = pack_bf16(a3.z, a3.w);
        v16bf A = __builtin_bit_cast(v16bf, Au);

        // B: 16 bf16 = two b128 loads, zero conversion VALU.
        u4 b0 = *(const u4*)(brow + k0 + koff);
        u4 b1 = *(const u4*)(brow + k0 + 16 + koff);
        v8u Bu = {b0.x, b0.y, b0.z, b0.w, b1.x, b1.y, b1.z, b1.w};
        v16bf B = __builtin_bit_cast(v16bf, Bu);

        // 8 args: (neg_a, A, neg_b, B, c_mod, C, reuse_a, reuse_b)
        acc = __builtin_amdgcn_wmma_f32_16x16x32_bf16(
                  false, A, false, B, (short)0, acc, false, false);
    }

    // C/D layout: lanes 0-15 -> M = j, lanes 16-31 -> M = 8+j; N = lane & 15.
    const int n     = lane & 15;
    const int mBase = (lane & 16) ? 8 : 0;
#pragma unroll
    for (int j = 0; j < 8; ++j)
        aff[(size_t)(tok0 + mBase + j) * NUM_EXPERT + n] = acc[j];
}

// ---------------------------------------------------------------------------
// Kernel C: reset the per-expert slot counters (no state across calls).
// ---------------------------------------------------------------------------
__global__ void reset_caps_kernel(int* __restrict__ cap) {
    if (threadIdx.x < NUM_EXPERT) cap[threadIdx.x] = 0;
}

// ---------------------------------------------------------------------------
// Kernel D: greedy balanced assignment. Each token claims a slot with its
// best-scoring expert that still has capacity (preference order). Total
// capacity == N and a counter only passes 1024 when that expert is truly
// full, so every token is guaranteed a slot.
// ---------------------------------------------------------------------------
__global__ __launch_bounds__(256)
void assign_kernel(const float* __restrict__ aff,
                   int* __restrict__ cap,
                   int* __restrict__ outIdx,
                   float* __restrict__ outVal) {
    const int t = blockIdx.x * blockDim.x + threadIdx.x;
    if (t >= N_TOKENS) return;

    const f4* r = (const f4*)(aff + (size_t)t * NUM_EXPERT);
    f4 s0 = r[0], s1 = r[1], s2 = r[2], s3 = r[3];
    float sc[16] = {s0.x, s0.y, s0.z, s0.w, s1.x, s1.y, s1.z, s1.w,
                    s2.x, s2.y, s2.z, s2.w, s3.x, s3.y, s3.z, s3.w};

    unsigned tried = 0;
    for (int attempt = 0; attempt < NUM_EXPERT; ++attempt) {
        int be = 0; float bv = -3.4e38f;
#pragma unroll
        for (int e = 0; e < NUM_EXPERT; ++e) {
            bool better = (((tried >> e) & 1u) == 0u) && (sc[e] > bv);
            bv = better ? sc[e] : bv;
            be = better ? e     : be;
        }
        tried |= (1u << be);
        int pos = atomicAdd(&cap[be], 1);
        if (pos < CAPACITY) {
            outIdx[be * CAPACITY + pos] = t;       // top_idx region
            outVal[be * CAPACITY + pos] = sc[be];  // top_value region
            return;
        }
    }
}

// ---------------------------------------------------------------------------
extern "C" void kernel_launch(void* const* d_in, const int* in_sizes, int n_in,
                              void* d_out, int out_size, void* d_ws, size_t ws_size,
                              hipStream_t stream) {
    const float* X    = (const float*)d_in[0];  // [4,4096,2048] -> [16384,2048]
    const float* Cent = (const float*)d_in[1];  // [16,2048]

    // Workspace layout: aff (1 MiB) | cap (64 B, padded to 256) | centbf (64 KiB)
    float* aff = (float*)d_ws;
    char*  p   = (char*)d_ws + (size_t)N_TOKENS * NUM_EXPERT * sizeof(float);
    int*   cap = (int*)p;
    unsigned* centbf = (unsigned*)(p + 256);

    int*   outIdx = (int*)d_out;                            // [16,1024] idx
    float* outVal = (float*)d_out + NUM_EXPERT * CAPACITY;  // [16,1024] val

    reset_caps_kernel<<<dim3(1), dim3(32), 0, stream>>>(cap);

    centroid_bf16_kernel<<<dim3((NUM_EXPERT * D_MODEL / 2 + 255) / 256),
                           dim3(256), 0, stream>>>(Cent, centbf);

    // 1024 tiles of 16 tokens, 8 waves (256 threads) per block -> 128 blocks.
    affinity_wmma_kernel<<<dim3(N_TOKENS / 16 / 8), dim3(256), 0, stream>>>(
        X, (const unsigned short*)centbf, aff);

    assign_kernel<<<dim3(N_TOKENS / 256), dim3(256), 0, stream>>>(
        aff, cap, outIdx, outVal);
}
